// VAE_1202590843594
// MI455X (gfx1250) — matrix operand order repaired and checked by reading, not compiled, not run on hardware
//
#include <hip/hip_runtime.h>

// Problem constants (from reference): S=128, B=32, I=1024, H=2048, G=4H
#define S_LEN  128
#define BATCH  32
#define IN_DIM 1024
#define HID    2048

typedef __bf16        bf16x16 __attribute__((ext_vector_type(16)));
typedef float         floatx8 __attribute__((ext_vector_type(8)));
typedef float         floatx4 __attribute__((ext_vector_type(4)));
typedef unsigned int  uintx4  __attribute__((ext_vector_type(4)));

union Frag {           // one WMMA 16-bit operand fragment: 16 bf16 per lane = 32B
  bf16x16 v;
  uintx4  u[2];
};

__device__ __forceinline__ unsigned short f32_bf16(float f) {
  unsigned int x = __float_as_uint(f);
  x += 0x7FFFu + ((x >> 16) & 1u);        // round-to-nearest-even
  return (unsigned short)(x >> 16);
}
__device__ __forceinline__ float sigmoidf_(float x) {
  return 1.0f / (1.0f + __expf(-x));
}

// Accumulate outT(16 units x 32 batch) += Wtile(16xK) * XT(Kx32) with bf16 WMMA.
// A-layout (16-bit A 16x32): lane l+16*hi holds row M=l, K = kb+16*hi .. kb+16*hi+15 (contiguous).
// B-layout (16-bit B 32x16): lane l+16*hi holds col N=l, K = kb+16*hi .. kb+16*hi+15 (contiguous).
__device__ __forceinline__ void mm_accum(
    const unsigned short* __restrict__ wrow,   // &W[(jbase+l)*K]  (this lane's weight row)
    const unsigned short* __restrict__ x0,     // &X[l*K]          (batch l)
    const unsigned short* __restrict__ x1,     // &X[(16+l)*K]     (batch 16+l)
    int K, int hi, floatx8& acc0, floatx8& acc1)
{
  for (int kb = 0; kb < K; kb += 32) {
    Frag a, b0, b1;
    const unsigned short* wp = wrow + kb + 16 * hi;
    a.u[0] = *(const uintx4*)(wp);
    a.u[1] = *(const uintx4*)(wp + 8);
    const unsigned short* p0 = x0 + kb + 16 * hi;
    b0.u[0] = *(const uintx4*)(p0);
    b0.u[1] = *(const uintx4*)(p0 + 8);
    const unsigned short* p1 = x1 + kb + 16 * hi;
    b1.u[0] = *(const uintx4*)(p1);
    b1.u[1] = *(const uintx4*)(p1 + 8);
    __builtin_prefetch((const void*)(wp + 64), 0, 1);   // next weight chunk -> global_prefetch_b8
    acc0 = __builtin_amdgcn_wmma_f32_16x16x32_bf16(false, a.v, false, b0.v, (short)0, acc0, false, false);
    acc1 = __builtin_amdgcn_wmma_f32_16x16x32_bf16(false, a.v, false, b1.v, (short)0, acc1, false, false);
  }
}

// Fused LSTM cell step: gates = xin*Wih^T + hprev*Whh^T + bih + bhh, then
// c = sig(f)*c + sig(i)*tanh(g); h = sig(o)*tanh(c).
// Grid: H/16 blocks (one 16-unit slice each). Block: 128 threads = 4 waves = gates i,f,g,o.
__global__ __launch_bounds__(128) void lstm_cell_kernel(
    const unsigned short* __restrict__ xin,  // (B, K1) bf16
    const unsigned short* __restrict__ Wih,  // (4H, K1) bf16
    int K1,
    const unsigned short* __restrict__ hprev,// (B, H) bf16
    const unsigned short* __restrict__ Whh,  // (4H, H) bf16
    const float* __restrict__ bih, const float* __restrict__ bhh,   // (4H)
    float* __restrict__ c,                   // (B, H) f32, in/out
    unsigned short* __restrict__ hout)       // (B, H) bf16
{
  const int tid  = threadIdx.x;
  const int gate = tid >> 5;        // 0=i 1=f 2=g 3=o
  const int lane = tid & 31;
  const int l    = lane & 15;
  const int hi   = lane >> 4;
  const int n16  = blockIdx.x * 16; // unit-tile base within H
  const int jrow = gate * HID + n16 + l;

  floatx8 acc0 = {};                // units (r+8*hi), batches 0..15
  floatx8 acc1 = {};                // units (r+8*hi), batches 16..31

  mm_accum(Wih + (size_t)jrow * K1,
           xin + (size_t)l * K1,
           xin + (size_t)(16 + l) * K1, K1, hi, acc0, acc1);
  mm_accum(Whh + (size_t)jrow * HID,
           hprev + (size_t)l * HID,
           hprev + (size_t)(16 + l) * HID, HID, hi, acc0, acc1);

  // biases: this lane owns units jb..jb+7
  const int jb = gate * HID + n16 + 8 * hi;
  floatx4 bi0 = *(const floatx4*)(bih + jb);
  floatx4 bi1 = *(const floatx4*)(bih + jb + 4);
  floatx4 bh0 = *(const floatx4*)(bhh + jb);
  floatx4 bh1 = *(const floatx4*)(bhh + jb + 4);
  floatx8 bb;
  bb[0]=bi0[0]+bh0[0]; bb[1]=bi0[1]+bh0[1]; bb[2]=bi0[2]+bh0[2]; bb[3]=bi0[3]+bh0[3];
  bb[4]=bi1[0]+bh1[0]; bb[5]=bi1[1]+bh1[1]; bb[6]=bi1[2]+bh1[2]; bb[7]=bi1[3]+bh1[3];
  acc0 += bb;
  acc1 += bb;

  // exchange the four gate tiles through LDS: g4[gate][batch][unit]
  __shared__ float g4[4][BATCH][16];
  *(floatx4*)&g4[gate][l][8*hi]          = __builtin_shufflevector(acc0, acc0, 0, 1, 2, 3);
  *(floatx4*)&g4[gate][l][8*hi + 4]      = __builtin_shufflevector(acc0, acc0, 4, 5, 6, 7);
  *(floatx4*)&g4[gate][16 + l][8*hi]     = __builtin_shufflevector(acc1, acc1, 0, 1, 2, 3);
  *(floatx4*)&g4[gate][16 + l][8*hi + 4] = __builtin_shufflevector(acc1, acc1, 4, 5, 6, 7);
  __syncthreads();

  for (int e = tid; e < BATCH * 16; e += 128) {
    int b = e >> 4, u = e & 15;
    float gi = g4[0][b][u];
    float gf = g4[1][b][u];
    float gg = g4[2][b][u];
    float go = g4[3][b][u];
    int idx = b * HID + n16 + u;
    float cn = sigmoidf_(gf) * c[idx] + sigmoidf_(gi) * tanhf(gg);
    c[idx] = cn;
    hout[idx] = f32_bf16(sigmoidf_(go) * tanhf(cn));
  }
}

// Generic out(M,N) = A(M,K) * W(N,K)^T + bias, f32 out, bf16 in.
// Grid: (M/32, N/64); block 128 (4 waves, one 16-col tile each).
// flipS: remap output row m=(t*B+b) -> ((S-1-t)*B+b) for the reversed x_hat.
__global__ __launch_bounds__(128) void gemm_bt_kernel(
    const unsigned short* __restrict__ A,
    const unsigned short* __restrict__ W,
    const float* __restrict__ bias,
    float* __restrict__ out,
    int M, int N, int K, int flipS)
{
  const int tid  = threadIdx.x;
  const int wave = tid >> 5;
  const int lane = tid & 31;
  const int l    = lane & 15;
  const int hi   = lane >> 4;
  const int m0   = blockIdx.x * 32;
  const int n0   = blockIdx.y * 64 + wave * 16;

  floatx8 acc0 = {};   // rows m0..m0+15
  floatx8 acc1 = {};   // rows m0+16..m0+31
  mm_accum(W + (size_t)(n0 + l) * K,
           A + (size_t)(m0 + l) * K,
           A + (size_t)(m0 + 16 + l) * K, K, hi, acc0, acc1);

  floatx4 b0 = *(const floatx4*)(bias + n0 + 8 * hi);
  floatx4 b1 = *(const floatx4*)(bias + n0 + 8 * hi + 4);
  floatx8 bb;
  bb[0]=b0[0]; bb[1]=b0[1]; bb[2]=b0[2]; bb[3]=b0[3];
  bb[4]=b1[0]; bb[5]=b1[1]; bb[6]=b1[2]; bb[7]=b1[3];
  acc0 += bb;
  acc1 += bb;

  int m_a = m0 + l;
  int m_b = m0 + 16 + l;
  if (flipS) {
    int t = m_a >> 5, b = m_a & 31;  m_a = (S_LEN - 1 - t) * BATCH + b;
    t = m_b >> 5;     b = m_b & 31;  m_b = (S_LEN - 1 - t) * BATCH + b;
  }
  float* o0 = out + (size_t)m_a * N + n0 + 8 * hi;
  *(floatx4*)(o0)     = __builtin_shufflevector(acc0, acc0, 0, 1, 2, 3);
  *(floatx4*)(o0 + 4) = __builtin_shufflevector(acc0, acc0, 4, 5, 6, 7);
  float* o1 = out + (size_t)m_b * N + n0 + 8 * hi;
  *(floatx4*)(o1)     = __builtin_shufflevector(acc1, acc1, 0, 1, 2, 3);
  *(floatx4*)(o1 + 4) = __builtin_shufflevector(acc1, acc1, 4, 5, 6, 7);
}

__global__ void cvt_bf16_kernel(const float* __restrict__ in, unsigned short* __restrict__ out, int n) {
  for (int i = blockIdx.x * blockDim.x + threadIdx.x; i < n; i += gridDim.x * blockDim.x)
    out[i] = f32_bf16(in[i]);
}
__global__ void zero_u16_kernel(unsigned short* p, int n) {
  for (int i = blockIdx.x * blockDim.x + threadIdx.x; i < n; i += gridDim.x * blockDim.x) p[i] = 0;
}
__global__ void zero_f32_kernel(float* p, int n) {
  for (int i = blockIdx.x * blockDim.x + threadIdx.x; i < n; i += gridDim.x * blockDim.x) p[i] = 0.0f;
}
// h = mu + eps * exp(0.5*logvar) -> decoder initial hidden (bf16) for layers 0 and 1
__global__ void reparam_kernel(const float* __restrict__ mu, const float* __restrict__ lv,
                               const float* __restrict__ eps,
                               unsigned short* __restrict__ h0, unsigned short* __restrict__ h1) {
  const int n = 2 * BATCH * HID;
  for (int i = blockIdx.x * blockDim.x + threadIdx.x; i < n; i += gridDim.x * blockDim.x) {
    float v = mu[i] + eps[i] * __expf(0.5f * lv[i]);
    if (i < BATCH * HID) h0[i] = f32_bf16(v);
    else                 h1[i - BATCH * HID] = f32_bf16(v);
  }
}

extern "C" void kernel_launch(void* const* d_in, const int* in_sizes, int n_in,
                              void* d_out, int out_size, void* d_ws, size_t ws_size,
                              hipStream_t stream)
{
  (void)in_sizes; (void)n_in; (void)out_size; (void)ws_size;
  const float* x     = (const float*)d_in[0];
  const float* eps   = (const float*)d_in[1];
  const float* eWih0 = (const float*)d_in[2];
  const float* eWhh0 = (const float*)d_in[3];
  const float* ebih0 = (const float*)d_in[4];
  const float* ebhh0 = (const float*)d_in[5];
  const float* eWih1 = (const float*)d_in[6];
  const float* eWhh1 = (const float*)d_in[7];
  const float* ebih1 = (const float*)d_in[8];
  const float* ebhh1 = (const float*)d_in[9];
  const float* muW   = (const float*)d_in[10];
  const float* mub   = (const float*)d_in[11];
  const float* lvW   = (const float*)d_in[12];
  const float* lvb   = (const float*)d_in[13];
  const float* dWih0 = (const float*)d_in[14];
  const float* dWhh0 = (const float*)d_in[15];
  const float* dbih0 = (const float*)d_in[16];
  const float* dbhh0 = (const float*)d_in[17];
  const float* dWih1 = (const float*)d_in[18];
  const float* dWhh1 = (const float*)d_in[19];
  const float* dbih1 = (const float*)d_in[20];
  const float* dbhh1 = (const float*)d_in[21];
  const float* recW  = (const float*)d_in[22];
  const float* recb  = (const float*)d_in[23];

  // ---- workspace bump allocator (256B aligned) ----
  char* wsb = (char*)d_ws;
  size_t off = 0;
  auto alloc_us = [&](size_t n) -> unsigned short* {
    unsigned short* p = (unsigned short*)(wsb + off);
    off = (off + n * 2 + 255) & ~(size_t)255;
    return p;
  };
  auto alloc_f = [&](size_t n) -> float* {
    float* p = (float*)(wsb + off);
    off = (off + n * 4 + 255) & ~(size_t)255;
    return p;
  };

  const size_t BH = (size_t)BATCH * HID;
  const size_t G  = (size_t)4 * HID;
  unsigned short* xb     = alloc_us((size_t)S_LEN * BATCH * IN_DIM);
  unsigned short* Wih0b  = alloc_us(G * IN_DIM);
  unsigned short* Whh0b  = alloc_us(G * HID);
  unsigned short* Wih1b  = alloc_us(G * HID);
  unsigned short* Whh1b  = alloc_us(G * HID);
  unsigned short* dWih0b = alloc_us(G * HID);
  unsigned short* dWhh0b = alloc_us(G * HID);
  unsigned short* dWih1b = alloc_us(G * HID);
  unsigned short* dWhh1b = alloc_us(G * HID);
  unsigned short* muWb   = alloc_us((size_t)HID * HID);
  unsigned short* lvWb   = alloc_us((size_t)HID * HID);
  unsigned short* recWb  = alloc_us((size_t)IN_DIM * HID);
  unsigned short* ys0    = alloc_us((size_t)(S_LEN + 1) * BH);  // enc L0 h, slot t+1 = step t
  unsigned short* ys1    = alloc_us((size_t)(S_LEN + 1) * BH);  // enc L1 h
  unsigned short* dh0    = alloc_us((size_t)(S_LEN + 1) * BH);  // dec L0 h
  unsigned short* dh1    = alloc_us((size_t)(S_LEN + 1) * BH);  // dec L1 h (slots 1..S = hs)
  unsigned short* xz     = alloc_us(BH);                        // zero decoder x0
  float* c0 = alloc_f(BH);
  float* c1 = alloc_f(BH);

  auto cvt = [&](const float* in, unsigned short* outp, size_t n) {
    int blocks = (int)((n + 2047) / 2048); if (blocks > 4096) blocks = 4096;
    cvt_bf16_kernel<<<blocks, 256, 0, stream>>>(in, outp, (int)n);
  };
  cvt(x,     xb,     (size_t)S_LEN * BATCH * IN_DIM);
  cvt(eWih0, Wih0b,  G * IN_DIM);
  cvt(eWhh0, Whh0b,  G * HID);
  cvt(eWih1, Wih1b,  G * HID);
  cvt(eWhh1, Whh1b,  G * HID);
  cvt(dWih0, dWih0b, G * HID);
  cvt(dWhh0, dWhh0b, G * HID);
  cvt(dWih1, dWih1b, G * HID);
  cvt(dWhh1, dWhh1b, G * HID);
  cvt(muW,   muWb,   (size_t)HID * HID);
  cvt(lvW,   lvWb,   (size_t)HID * HID);
  cvt(recW,  recWb,  (size_t)IN_DIM * HID);

  zero_u16_kernel<<<64, 256, 0, stream>>>(ys0, (int)BH);  // h0 init = 0
  zero_u16_kernel<<<64, 256, 0, stream>>>(ys1, (int)BH);
  zero_u16_kernel<<<64, 256, 0, stream>>>(xz,  (int)BH);
  zero_f32_kernel<<<64, 256, 0, stream>>>(c0,  (int)BH);  // c0 init = 0
  zero_f32_kernel<<<64, 256, 0, stream>>>(c1,  (int)BH);

  float* out_x  = (float*)d_out;
  float* out_mu = out_x + (size_t)S_LEN * BATCH * IN_DIM;
  float* out_lv = out_mu + 2 * BH;

  // ---- encoder: 2-layer LSTM, 128 sequential steps ----
  for (int t = 0; t < S_LEN; ++t) {
    lstm_cell_kernel<<<HID / 16, 128, 0, stream>>>(
        xb + (size_t)t * BATCH * IN_DIM, Wih0b, IN_DIM,
        ys0 + (size_t)t * BH, Whh0b, ebih0, ebhh0,
        c0, ys0 + (size_t)(t + 1) * BH);
    lstm_cell_kernel<<<HID / 16, 128, 0, stream>>>(
        ys0 + (size_t)(t + 1) * BH, Wih1b, HID,
        ys1 + (size_t)t * BH, Whh1b, ebih1, ebhh1,
        c1, ys1 + (size_t)(t + 1) * BH);
  }

  // ---- mu / logvar (M=32 each layer) ----
  dim3 gmu(1, HID / 64);
  gemm_bt_kernel<<<gmu, 128, 0, stream>>>(ys0 + (size_t)S_LEN * BH, muWb, mub, out_mu,      32, HID, HID, 0);
  gemm_bt_kernel<<<gmu, 128, 0, stream>>>(ys1 + (size_t)S_LEN * BH, muWb, mub, out_mu + BH, 32, HID, HID, 0);
  gemm_bt_kernel<<<gmu, 128, 0, stream>>>(ys0 + (size_t)S_LEN * BH, lvWb, lvb, out_lv,      32, HID, HID, 0);
  gemm_bt_kernel<<<gmu, 128, 0, stream>>>(ys1 + (size_t)S_LEN * BH, lvWb, lvb, out_lv + BH, 32, HID, HID, 0);

  // ---- reparameterize -> decoder initial hidden states (slot 0); c carries over ----
  reparam_kernel<<<512, 256, 0, stream>>>(out_mu, out_lv, eps, dh0, dh1);

  // ---- decoder: autoregressive 2-layer LSTM ----
  for (int t = 0; t < S_LEN; ++t) {
    const unsigned short* xprev = (t == 0) ? xz : (dh1 + (size_t)t * BH);
    lstm_cell_kernel<<<HID / 16, 128, 0, stream>>>(
        xprev, dWih0b, HID,
        dh0 + (size_t)t * BH, dWhh0b, dbih0, dbhh0,
        c0, dh0 + (size_t)(t + 1) * BH);
    lstm_cell_kernel<<<HID / 16, 128, 0, stream>>>(
        dh0 + (size_t)(t + 1) * BH, dWih1b, HID,
        dh1 + (size_t)t * BH, dWhh1b, dbih1, dbhh1,
        c1, dh1 + (size_t)(t + 1) * BH);
  }

  // ---- readout over all steps at once, flip folded into store index ----
  dim3 gro((S_LEN * BATCH) / 32, IN_DIM / 64);
  gemm_bt_kernel<<<gro, 128, 0, stream>>>(dh1 + BH, recWb, recb, out_x,
                                          S_LEN * BATCH, IN_DIM, HID, 1);
}
